// HiLoAttention_16423954940066
// MI455X (gfx1250) — matrix operand order repaired and checked by reading, not compiled, not run on hardware
//
#include <hip/hip_runtime.h>

// ---------------------------------------------------------------------------
// HiLo attention for MI455X (gfx1250, wave32, WMMA + Tensor Data Mover).
// fp16 data path with f32 accumulation via v_wmma_f32_16x16x32_f16.
// GEMMs stage A/B tiles in LDS via TDM (tensor_load_to_lds, TENSORcnt).
// ---------------------------------------------------------------------------

typedef _Float16 f16;
typedef __attribute__((ext_vector_type(16))) _Float16 v16h;
typedef __attribute__((ext_vector_type(8)))  _Float16 v8h;
typedef __attribute__((ext_vector_type(8)))  float    v8f;
typedef unsigned int __attribute__((ext_vector_type(4))) uint4v;
typedef int __attribute__((ext_vector_type(8)))          int8v;
typedef int __attribute__((ext_vector_type(4)))          int4v;

#define DEV static __device__ __forceinline__

// ---- fixed problem shape ---------------------------------------------------
constexpr int B_    = 2;
constexpr int HDIM  = 64, WDIM = 64;
constexpr int C_    = 1024;
constexpr int N_    = HDIM * WDIM;       // 4096 tokens per batch
constexpr int HD    = 32;                // head dim
constexpr int LC    = 512, HC = 512;
constexpr int WS    = 2;
constexpr int NH2   = HDIM / WS, NW2 = WDIM / WS;   // 32 x 32
constexpr int NPOOL = NH2 * NW2;         // 1024 pooled tokens per batch
constexpr int ROWS  = B_ * N_;           // 8192
constexpr int PROWS = B_ * NPOOL;        // 2048
constexpr float SCALE = 0.17677669529663687f;   // 32^-0.5

// ---- workspace layout (bytes) ---------------------------------------------
constexpr size_t OFF_XH    = 0;                                   // 8192x1024 f16
constexpr size_t OFF_XP    = OFF_XH    + (size_t)ROWS * C_ * 2;   // 2048x1024 f16
constexpr size_t OFF_WHQKV = OFF_XP    + (size_t)PROWS * C_ * 2;  // 1536x1024 f16 (transposed)
constexpr size_t OFF_WLQ   = OFF_WHQKV + (size_t)1536 * 1024 * 2; // 512x1024
constexpr size_t OFF_WLKV  = OFF_WLQ   + (size_t)512 * 1024 * 2;  // 1024x1024
constexpr size_t OFF_WHP   = OFF_WLKV  + (size_t)1024 * 1024 * 2; // 512x512
constexpr size_t OFF_WLP   = OFF_WHP   + (size_t)512 * 512 * 2;   // 512x512
constexpr size_t OFF_QKV   = OFF_WLP   + (size_t)512 * 512 * 2;   // 8192x1536 f16
constexpr size_t OFF_Q     = OFF_QKV   + (size_t)ROWS * 1536 * 2; // 8192x512
constexpr size_t OFF_KV    = OFF_Q     + (size_t)ROWS * 512 * 2;  // 2048x1024
constexpr size_t OFF_OH    = OFF_KV    + (size_t)PROWS * 1024 * 2;// 8192x512
constexpr size_t OFF_OL    = OFF_OH    + (size_t)ROWS * 512 * 2;  // 8192x512

// ---------------------------------------------------------------------------
// WMMA fragment helpers (A/B: contiguous k order, not ISA table order; C/D per CDNA5 ISA 7.12.2)
// A 16x32 f16: lane m=l&15, g=l>>4; elems 0..15 -> K=16g+0..15
// B 32x16 f16: lane n=l&15, g=l>>4; elems 0..15 -> K=16g+0..15
// C/D 16x16 f32: VGPR r, lane l -> row r+8*(l>>4), col l&15
// ---------------------------------------------------------------------------
DEV v16h load_a_frag(const f16* __restrict__ base, int ld, int m0, int k0, int lane) {
  const int m = m0 + (lane & 15);
  const int g = lane >> 4;
  const f16* p = base + (size_t)m * ld + k0 + 16 * g;
  v8h lo = *(const v8h*)p;
  v8h hi = *(const v8h*)(p + 8);
  v16h a;
#pragma unroll
  for (int i = 0; i < 8; ++i) { a[i] = lo[i]; a[i + 8] = hi[i]; }
  return a;
}

DEV v16h load_b_frag(const f16* __restrict__ base, int ld, int n0, int k0, int lane) {
  const int n = n0 + (lane & 15);
  const int g = lane >> 4;
  const f16* p = base + (size_t)n * ld + k0 + 16 * g;
  v8h lo = *(const v8h*)p;
  v8h hi = *(const v8h*)(p + 8);
  v16h b;
#pragma unroll
  for (int i = 0; i < 8; ++i) { b[i] = lo[i]; b[i + 8] = hi[i]; }
  return b;
}

DEV v8f wmma_f16(v16h a, v16h b, v8f c) {
  return __builtin_amdgcn_wmma_f32_16x16x32_f16(false, a, false, b, (short)0, c,
                                                false, false);
}

// ---------------------------------------------------------------------------
// TDM: DMA one 2-D f16 tile [rows x cols] (row stride ld elems) into LDS.
// D# packed per CDNA5 ISA 8.3/8.4: group0 {count=1, lds_addr, global_addr,
// type=2}; group1 {data_size=2B, tensor_dim0/1 = tile dims, tile_dim0/1,
// tensor_dim0_stride = ld}. Groups 2/3 zero (2-D tensor); trailing int32x8
// slot zero (6-arg toolchain form). Issue once per wave.
// ---------------------------------------------------------------------------
DEV void tdm_load_tile_f16(const f16* g, unsigned lds_byte_off,
                           int rows, int cols, int ld) {
  unsigned long long ga = (unsigned long long)(size_t)g;
  uint4v g0 = {};
  g0[0] = 1u;                                             // count=1, load, user
  g0[1] = lds_byte_off;                                   // lds_addr (bytes)
  g0[2] = (unsigned)ga;                                   // global_addr[31:0]
  g0[3] = ((unsigned)(ga >> 32) & 0x01ffffffu) | (2u << 30);  // addr[56:32],type=2
  int8v g1 = {};
  g1[0] = (int)(1u << 16);                                // data_size = 2 bytes
  g1[1] = (int)(((unsigned)cols & 0xffffu) << 16);        // tensor_dim0[15:0]
  g1[2] = (int)((((unsigned)cols >> 16) & 0xffffu) |
                (((unsigned)rows & 0xffffu) << 16));      // dim0 hi / dim1 lo
  g1[3] = (int)((((unsigned)rows >> 16) & 0xffffu) |
                (((unsigned)cols & 0xffffu) << 16));      // dim1 hi / tile_dim0
  g1[4] = (int)((unsigned)rows & 0xffffu);                // tile_dim1 (tile_dim2=0)
  g1[5] = ld;                                             // tensor_dim0_stride lo
  g1[6] = 0;
  g1[7] = 0;
  int4v z4 = (int4v)0;
  int8v z8 = (int8v)0;
  __builtin_amdgcn_tensor_load_to_lds(g0, g1, z4, z4, z8, 0);
}

// ---------------------------------------------------------------------------
// small prep kernels
// ---------------------------------------------------------------------------
__global__ void cast_f32_f16_kernel(const float* __restrict__ in,
                                    f16* __restrict__ out, int n) {
  int i = blockIdx.x * blockDim.x + threadIdx.x;
  if (i < n) out[i] = (f16)in[i];
}

// W: [K,N] f32 row-major -> Wt: [N,K] f16 row-major (coalesced reads)
__global__ void transpose_cast_kernel(const float* __restrict__ W,
                                      f16* __restrict__ Wt, int K, int N) {
  int i = blockIdx.x * blockDim.x + threadIdx.x;
  if (i >= K * N) return;
  int n = i % N, k = i / N;
  Wt[(size_t)n * K + k] = (f16)W[i];
}

// 2x2 average pool of x[B,H,W,C] -> xp[B*NPOOL, C] f16
__global__ void pool_kernel(const float* __restrict__ x, f16* __restrict__ xp) {
  int i = blockIdx.x * blockDim.x + threadIdx.x;
  if (i >= PROWS * C_) return;
  int c = i % C_;
  int p = i / C_;
  int b = p / NPOOL;
  int pr = p % NPOOL;
  int wh = pr / NW2, ww = pr % NW2;
  const float* xb = x + (size_t)b * N_ * C_;
  float s = 0.f;
#pragma unroll
  for (int ii = 0; ii < WS; ++ii)
#pragma unroll
    for (int jj = 0; jj < WS; ++jj)
      s += xb[(size_t)((2 * wh + ii) * WDIM + (2 * ww + jj)) * C_ + c];
  xp[i] = (f16)(0.25f * s);
}

// ---------------------------------------------------------------------------
// GEMM: out[M,N] = A[M,K](f16) @ Wt[N,K]^T(f16) + bias (f32 accumulate).
// 256 threads = 8 waves (4 along M x 2 along N); WG tile 128x64; wave 32x32.
// A/B tiles double-buffered in dynamic LDS, filled by TDM issued from wave 0,
// synchronized with s_wait_tensorcnt + workgroup barrier; DMA of step k+1
// overlaps WMMA compute of step k.
// Dynamic LDS layout (f16 elems): A0[0,4096) A1[4096,8192) B0[8192,10240)
//                                 B1[10240,12288)  -> 24576 bytes.
// ---------------------------------------------------------------------------
constexpr unsigned GEMM_LDS_BYTES = (2 * 128 * 32 + 2 * 64 * 32) * 2;

__global__ __launch_bounds__(256) void gemm_f16_kernel(
    const f16* __restrict__ A, const f16* __restrict__ Wt,
    const float* __restrict__ bias, float* __restrict__ outF,
    f16* __restrict__ outH, int K, int ldOut, int colOff) {
  extern __shared__ f16 smem[];
  const int lane = threadIdx.x & 31;
  const int wave = threadIdx.x >> 5;
  const int wm = wave & 3;          // 0..3 along M
  const int wn = wave >> 2;         // 0..1 along N
  const int mBase = blockIdx.y * 128;
  const int nBase = blockIdx.x * 64;
  const int nk = K >> 5;

  if (wave == 0) {   // prologue: DMA first K-slab
    tdm_load_tile_f16(A + (size_t)mBase * K, 0u, 128, 32, K);
    tdm_load_tile_f16(Wt + (size_t)nBase * K, 8192u * 2u, 64, 32, K);
  }

  v8f acc[2][2] = {};
  for (int ks = 0; ks < nk; ++ks) {
    const int cur = ks & 1;
    if (wave == 0) __builtin_amdgcn_s_wait_tensorcnt(0);
    __syncthreads();                          // buf[cur] ready for all waves
    if (wave == 0 && ks + 1 < nk) {           // overlap DMA of next slab
      const int nxt = cur ^ 1;
      tdm_load_tile_f16(A + (size_t)mBase * K + (ks + 1) * 32,
                        (unsigned)(nxt * 4096) * 2u, 128, 32, K);
      tdm_load_tile_f16(Wt + (size_t)nBase * K + (ks + 1) * 32,
                        (unsigned)(8192 + nxt * 2048) * 2u, 64, 32, K);
    }
    const f16* sA = smem + cur * 4096;        // [128][32]
    const f16* sB = smem + 8192 + cur * 2048; // [64][32]
    v16h a0 = load_a_frag(sA, 32, wm * 32, 0, lane);
    v16h a1 = load_a_frag(sA, 32, wm * 32 + 16, 0, lane);
    v16h b0 = load_b_frag(sB, 32, wn * 32, 0, lane);
    v16h b1 = load_b_frag(sB, 32, wn * 32 + 16, 0, lane);
    acc[0][0] = wmma_f16(a0, b0, acc[0][0]);
    acc[0][1] = wmma_f16(a0, b1, acc[0][1]);
    acc[1][0] = wmma_f16(a1, b0, acc[1][0]);
    acc[1][1] = wmma_f16(a1, b1, acc[1][1]);
    __syncthreads();                          // done reading buf[cur]
  }

  const int g = lane >> 4, nl = lane & 15;
  const int m0 = mBase + wm * 32;
  const int n0 = nBase + wn * 32;
#pragma unroll
  for (int fi = 0; fi < 2; ++fi)
#pragma unroll
    for (int fj = 0; fj < 2; ++fj) {
      const int ncol = n0 + 16 * fj + nl;
      const float bv = bias ? bias[ncol] : 0.f;
#pragma unroll
      for (int r = 0; r < 8; ++r) {
        const int row = m0 + 16 * fi + 8 * g + r;
        const float v = acc[fi][fj][r] + bv;
        if (outF) outF[(size_t)row * ldOut + colOff + ncol] = v;
        else      outH[(size_t)row * ldOut + colOff + ncol] = (f16)v;
      }
    }
}

// ---------------------------------------------------------------------------
// Hi-Fi: 2x2-window attention, one wave per (window, head).
// lane = head-dim index d (0..31); 4x4 logits via wave-wide shuffle reduce.
// ---------------------------------------------------------------------------
__global__ __launch_bounds__(256) void hifi_attn_kernel(
    const f16* __restrict__ qkv, f16* __restrict__ oh) {
  const int lane = threadIdx.x & 31;
  const int wid = (blockIdx.x * 256 + threadIdx.x) >> 5;  // 0..32767
  const int head = wid & 15;
  const int win = wid >> 4;        // 0..2047
  const int b = win >> 10;
  const int wr = win & 1023;
  const int wh = wr >> 5, ww = wr & 31;

  int t[4];
#pragma unroll
  for (int i = 0; i < 2; ++i)
#pragma unroll
    for (int j = 0; j < 2; ++j)
      t[i * 2 + j] = b * N_ + (2 * wh + i) * WDIM + (2 * ww + j);

  float q[4], k[4], v[4];
#pragma unroll
  for (int i = 0; i < 4; ++i) {
    const f16* pt = qkv + (size_t)t[i] * (3 * HC) + head * HD + lane;
    q[i] = (float)pt[0];
    k[i] = (float)pt[HC];
    v[i] = (float)pt[2 * HC];
  }

  float s[4][4];
#pragma unroll
  for (int i = 0; i < 4; ++i)
#pragma unroll
    for (int j = 0; j < 4; ++j) {
      float p = q[i] * k[j];
#pragma unroll
      for (int off = 16; off > 0; off >>= 1) p += __shfl_xor(p, off, 32);
      s[i][j] = p * SCALE;
    }

#pragma unroll
  for (int i = 0; i < 4; ++i) {
    float m = fmaxf(fmaxf(s[i][0], s[i][1]), fmaxf(s[i][2], s[i][3]));
    float e[4], sum = 0.f;
#pragma unroll
    for (int j = 0; j < 4; ++j) { e[j] = __expf(s[i][j] - m); sum += e[j]; }
    const float inv = 1.f / sum;
    float o = 0.f;
#pragma unroll
    for (int j = 0; j < 4; ++j) o += e[j] * inv * v[j];
    oh[(size_t)t[i] * HC + head * HD + lane] = (f16)o;
  }
}

// ---------------------------------------------------------------------------
// Lo-Fi: flash-style attention. Q[4096,32] vs pooled K,V[1024,32] per (b,head).
// WG = 4 waves; each wave owns a 16-row Q tile; K/V^T blocks (32 keys) staged
// in LDS cooperatively; S and P@V through WMMA; online softmax with
// half-wave shuffle reductions (C-layout: lane owns rows 8g+r).
// ---------------------------------------------------------------------------
__global__ __launch_bounds__(128) void lofi_attn_kernel(
    const f16* __restrict__ q, const f16* __restrict__ kv,
    f16* __restrict__ ol) {
  __shared__ __align__(16) f16 sK[32 * 32];        // [key][dim]
  __shared__ __align__(16) f16 sVT[32 * 32];       // [dim][key]
  __shared__ __align__(16) f16 sP[4][16 * 32];     // per-wave P staging

  const int lane = threadIdx.x & 31;
  const int wave = threadIdx.x >> 5;
  const int bh = blockIdx.x >> 6;       // 0..31 : (b, head)
  const int tg = blockIdx.x & 63;       // q-tile group
  const int b = bh >> 4, head = bh & 15;
  const int qrow0 = b * N_ + tg * 64 + wave * 16;
  const int kvrow0 = b * NPOOL;
  const int g = lane >> 4, nl = lane & 15;

  const v16h qa = load_a_frag(q + head * HD, LC, qrow0, 0, lane);

  v8f oc0 = {}, oc1 = {};
  float mrow[8], lrow[8];
#pragma unroll
  for (int r = 0; r < 8; ++r) { mrow[r] = -1e30f; lrow[r] = 0.f; }

  for (int j0 = 0; j0 < NPOOL; j0 += 32) {
    // cooperative stage: K block row-major, V block transposed
    {
      const int key = threadIdx.x >> 2;   // 0..31
      const int seg = threadIdx.x & 3;    // 0..3
      const f16* krow =
          kv + (size_t)(kvrow0 + j0 + key) * (2 * LC) + head * HD + seg * 8;
      *(v8h*)&sK[key * 32 + seg * 8] = *(const v8h*)krow;
      v8h vv = *(const v8h*)(krow + LC);
#pragma unroll
      for (int i = 0; i < 8; ++i) sVT[(seg * 8 + i) * 32 + key] = vv[i];
    }
    __syncthreads();

    // S = Q @ K^T  (two 16-key fragments)
    v16h bk0, bk1;
    {
      const f16* p0 = &sK[nl * 32 + 16 * g];
      const f16* p1 = &sK[(nl + 16) * 32 + 16 * g];
      v8h l0 = *(const v8h*)p0, h0 = *(const v8h*)(p0 + 8);
      v8h l1 = *(const v8h*)p1, h1 = *(const v8h*)(p1 + 8);
#pragma unroll
      for (int i = 0; i < 8; ++i) {
        bk0[i] = l0[i]; bk0[i + 8] = h0[i];
        bk1[i] = l1[i]; bk1[i + 8] = h1[i];
      }
    }
    v8f s0 = {}, s1 = {};
    s0 = wmma_f16(qa, bk0, s0);
    s1 = wmma_f16(qa, bk1, s1);

    // online softmax (per-lane rows 8g+r); reductions stay in 16-lane halves
    float mnew[8], fsc[8];
#pragma unroll
    for (int r = 0; r < 8; ++r) {
      s0[r] *= SCALE; s1[r] *= SCALE;
      float t = fmaxf(s0[r], s1[r]);
#pragma unroll
      for (int off = 1; off < 16; off <<= 1) t = fmaxf(t, __shfl_xor(t, off, 32));
      mnew[r] = fmaxf(mrow[r], t);
      fsc[r] = __expf(mrow[r] - mnew[r]);
      mrow[r] = mnew[r];
    }
#pragma unroll
    for (int r = 0; r < 8; ++r) {
      float p0 = __expf(s0[r] - mnew[r]);
      float p1 = __expf(s1[r] - mnew[r]);
      s0[r] = p0; s1[r] = p1;
      float t = p0 + p1;
#pragma unroll
      for (int off = 1; off < 16; off <<= 1) t += __shfl_xor(t, off, 32);
      lrow[r] = lrow[r] * fsc[r] + t;
      oc0[r] *= fsc[r];
      oc1[r] *= fsc[r];
    }

    // P: C-layout regs -> LDS -> A-layout fragment (register transpose via LDS)
    f16* P = sP[wave];
#pragma unroll
    for (int r = 0; r < 8; ++r) {
      P[(8 * g + r) * 32 + nl]      = (f16)s0[r];
      P[(8 * g + r) * 32 + 16 + nl] = (f16)s1[r];
    }
    v16h pa;
    {
      const f16* p = &P[nl * 32 + 16 * g];
      v8h lo = *(const v8h*)p, hi = *(const v8h*)(p + 8);
#pragma unroll
      for (int i = 0; i < 8; ++i) { pa[i] = lo[i]; pa[i + 8] = hi[i]; }
    }

    // O += P @ V  (dims split 0..15 / 16..31)
    v16h bv0, bv1;
    {
      const f16* p0 = &sVT[nl * 32 + 16 * g];
      const f16* p1 = &sVT[(16 + nl) * 32 + 16 * g];
      v8h l0 = *(const v8h*)p0, h0 = *(const v8h*)(p0 + 8);
      v8h l1 = *(const v8h*)p1, h1 = *(const v8h*)(p1 + 8);
#pragma unroll
      for (int i = 0; i < 8; ++i) {
        bv0[i] = l0[i]; bv0[i + 8] = h0[i];
        bv1[i] = l1[i]; bv1[i + 8] = h1[i];
      }
    }
    oc0 = wmma_f16(pa, bv0, oc0);
    oc1 = wmma_f16(pa, bv1, oc1);
    __syncthreads();
  }

#pragma unroll
  for (int r = 0; r < 8; ++r) {
    const float inv = 1.f / lrow[r];
    const int row = qrow0 + 8 * g + r;
    ol[(size_t)row * LC + head * HD + nl]      = (f16)(oc0[r] * inv);
    ol[(size_t)row * LC + head * HD + 16 + nl] = (f16)(oc1[r] * inv);
  }
}

// ---------------------------------------------------------------------------
extern "C" void kernel_launch(void* const* d_in, const int* in_sizes, int n_in,
                              void* d_out, int out_size, void* d_ws,
                              size_t ws_size, hipStream_t stream) {
  (void)in_sizes; (void)n_in; (void)out_size; (void)ws_size;
  const float* x        = (const float*)d_in[0];
  const float* h_qkv_w  = (const float*)d_in[1];
  const float* h_qkv_b  = (const float*)d_in[2];
  const float* h_proj_w = (const float*)d_in[3];
  const float* h_proj_b = (const float*)d_in[4];
  const float* l_q_w    = (const float*)d_in[5];
  const float* l_q_b    = (const float*)d_in[6];
  const float* l_kv_w   = (const float*)d_in[7];
  const float* l_kv_b   = (const float*)d_in[8];
  const float* l_proj_w = (const float*)d_in[9];
  const float* l_proj_b = (const float*)d_in[10];

  char* ws = (char*)d_ws;
  f16* xh    = (f16*)(ws + OFF_XH);
  f16* xp    = (f16*)(ws + OFF_XP);
  f16* wHqkv = (f16*)(ws + OFF_WHQKV);
  f16* wLq   = (f16*)(ws + OFF_WLQ);
  f16* wLkv  = (f16*)(ws + OFF_WLKV);
  f16* wHp   = (f16*)(ws + OFF_WHP);
  f16* wLp   = (f16*)(ws + OFF_WLP);
  f16* qkv16 = (f16*)(ws + OFF_QKV);
  f16* q16   = (f16*)(ws + OFF_Q);
  f16* kv16  = (f16*)(ws + OFF_KV);
  f16* oh16  = (f16*)(ws + OFF_OH);
  f16* ol16  = (f16*)(ws + OFF_OL);
  float* out = (float*)d_out;

  // 1) precision conversion / pooling / weight transposes
  {
    int n = ROWS * C_;
    cast_f32_f16_kernel<<<(n + 255) / 256, 256, 0, stream>>>(x, xh, n);
  }
  pool_kernel<<<(PROWS * C_ + 255) / 256, 256, 0, stream>>>(x, xp);
  transpose_cast_kernel<<<(1024 * 1536 + 255) / 256, 256, 0, stream>>>(h_qkv_w, wHqkv, 1024, 1536);
  transpose_cast_kernel<<<(1024 * 512  + 255) / 256, 256, 0, stream>>>(l_q_w,   wLq,   1024, 512);
  transpose_cast_kernel<<<(1024 * 1024 + 255) / 256, 256, 0, stream>>>(l_kv_w,  wLkv,  1024, 1024);
  transpose_cast_kernel<<<(512 * 512   + 255) / 256, 256, 0, stream>>>(h_proj_w, wHp,  512, 512);
  transpose_cast_kernel<<<(512 * 512   + 255) / 256, 256, 0, stream>>>(l_proj_w, wLp,  512, 512);

  const dim3 blk(256);
  // 2) input projections (f16 outputs), TDM-staged LDS tiles
  gemm_f16_kernel<<<dim3(1536 / 64, ROWS / 128), blk, GEMM_LDS_BYTES, stream>>>(
      xh, wHqkv, h_qkv_b, nullptr, qkv16, 1024, 1536, 0);
  gemm_f16_kernel<<<dim3(512 / 64, ROWS / 128), blk, GEMM_LDS_BYTES, stream>>>(
      xh, wLq, l_q_b, nullptr, q16, 1024, 512, 0);
  gemm_f16_kernel<<<dim3(1024 / 64, PROWS / 128), blk, GEMM_LDS_BYTES, stream>>>(
      xp, wLkv, l_kv_b, nullptr, kv16, 1024, 1024, 0);

  // 3) attention
  hifi_attn_kernel<<<4096, 256, 0, stream>>>(qkv16, oh16);   // 32768 waves
  lofi_attn_kernel<<<2048, 128, 0, stream>>>(q16, kv16, ol16);

  // 4) output projections straight into d_out (high channels first)
  gemm_f16_kernel<<<dim3(512 / 64, ROWS / 128), blk, GEMM_LDS_BYTES, stream>>>(
      oh16, wHp, h_proj_b, out, nullptr, 512, 1024, 0);
  gemm_f16_kernel<<<dim3(512 / 64, ROWS / 128), blk, GEMM_LDS_BYTES, stream>>>(
      ol16, wLp, l_proj_b, out, nullptr, 512, 1024, 512);
}
